// GroupedExperts_13864154432368
// MI455X (gfx1250) — matrix operand order repaired and checked by reading, not compiled, not run on hardware
//
#include <hip/hip_runtime.h>
#include <hip/hip_bf16.h>

// Problem constants (match reference)
#define T_TOK 2048
#define D_DIM 1024
#define I_DIM 1024
#define N_EXP 8
#define TOPK  2
#define RPE   (T_TOK * TOPK)   // worst-case routed rows per expert
#define TILE_M 32              // 2 WMMA M-subtiles per block: B-fragment reuse x2

typedef __attribute__((ext_vector_type(16))) __bf16 v16bf;
typedef __attribute__((ext_vector_type(8)))  float  v8f;
typedef __attribute__((ext_vector_type(4)))  unsigned int u32x4;

union FragAB { u32x4 u[2]; v16bf v; };   // 32 bytes = 8 VGPRs = one A/B fragment

__device__ __forceinline__ unsigned short f32_to_bf16(float f) {
    unsigned int u = __builtin_bit_cast(unsigned int, f);
    unsigned int r = u + 0x7FFFu + ((u >> 16) & 1u);   // round-to-nearest-even
    return (unsigned short)(r >> 16);
}

__device__ __forceinline__ float silu_mul(float g, float u) {
    return (g / (1.0f + __expf(-g))) * u;
}

// ---------------------------------------------------------------- zero counts
__global__ void zero_counts_kernel(int* counts) {
    if (threadIdx.x < N_EXP) counts[threadIdx.x] = 0;
}

// ---------------------------------------------------------------- routing
__global__ void route_kernel(const int* __restrict__ indices,
                             const float* __restrict__ weights,
                             const unsigned char* __restrict__ token_mask,
                             int* __restrict__ counts,
                             int* __restrict__ lists,
                             float* __restrict__ wl) {
    int t = blockIdx.x * blockDim.x + threadIdx.x;
    if (t >= T_TOK) return;
    float m = token_mask[t] ? 1.0f : 0.0f;
    for (int k = 0; k < TOPK; ++k) {
        int e = indices[t * TOPK + k];
        float w = weights[t * TOPK + k] * m;
        int pos = atomicAdd(&counts[e], 1);
        lists[e * RPE + pos] = t * TOPK + k;   // slot id
        wl[e * RPE + pos]    = w;
    }
}

// ---------------------------------------------------------------- fp32 -> bf16
__global__ void cvt_kernel(const float* __restrict__ src,
                           unsigned short* __restrict__ dst, int n4) {
    int i      = blockIdx.x * blockDim.x + threadIdx.x;
    int stride = gridDim.x * blockDim.x;
    for (int v = i; v < n4; v += stride) {
        const float4 f = reinterpret_cast<const float4*>(src)[v];
        unsigned int lo = ((unsigned int)f32_to_bf16(f.y) << 16) | f32_to_bf16(f.x);
        unsigned int hi = ((unsigned int)f32_to_bf16(f.w) << 16) | f32_to_bf16(f.z);
        reinterpret_cast<uint2*>(dst)[v] = make_uint2(lo, hi);
    }
}

// ---------------------------------------------------------------- main MoE GEMM
// Block = 256 threads (8 waves). Each block: one expert, one 32-row token tile
// (two 16-row WMMA subtiles -> every weight B-fragment feeds 2 WMMAs).
// Phase A: h[32, I] = silu(x Gt) * (x Ut)   (each wave owns 128 columns of I)
// Phase B: out[32, D] = h Dnt, scaled by routing weight, stored per slot.
__launch_bounds__(256)
__global__ void moe_kernel(const unsigned short* __restrict__ xb,
                           const unsigned short* __restrict__ gb,
                           const unsigned short* __restrict__ ub,
                           const unsigned short* __restrict__ db,
                           const int*   __restrict__ counts,
                           const int*   __restrict__ lists,
                           const float* __restrict__ wl,
                           float* __restrict__ outslot) {
    __shared__ int            s_slot[TILE_M];
    __shared__ float          s_w[TILE_M];
    __shared__ unsigned short s_h[TILE_M * I_DIM];   // 64 KB bf16 h tile

    const int e   = blockIdx.y;
    const int cnt = counts[e];
    const int m0  = blockIdx.x * TILE_M;
    if (m0 >= cnt) return;

    const int tid = threadIdx.x;
    if (tid < TILE_M) {
        int r = m0 + tid;
        if (r < cnt) { s_slot[tid] = lists[e * RPE + r]; s_w[tid] = wl[e * RPE + r]; }
        else         { s_slot[tid] = -1;                 s_w[tid] = 0.0f; }
    }
    __syncthreads();

    const int lane = tid & 31;
    const int wv   = tid >> 5;     // 0..7
    const int half = lane >> 4;    // 0/1
    const int lrow = lane & 15;

    const size_t ew = (size_t)e * I_DIM * D_DIM;

    // x rows for this lane's two A-matrix rows (M = lrow and lrow+16)
    int sa0 = s_slot[lrow];
    int sa1 = s_slot[lrow + 16];
    const unsigned short* xrow0 = xb + (size_t)(sa0 < 0 ? 0 : sa0 / TOPK) * D_DIM;
    const unsigned short* xrow1 = xb + (size_t)(sa1 < 0 ? 0 : sa1 / TOPK) * D_DIM;

    // ---------------- Phase A ----------------
    for (int nt = 0; nt < 8; ++nt) {
        const int n0 = wv * 128 + nt * 16;
        v8f aG0 = {0.f,0.f,0.f,0.f,0.f,0.f,0.f,0.f};
        v8f aG1 = {0.f,0.f,0.f,0.f,0.f,0.f,0.f,0.f};
        v8f aU0 = {0.f,0.f,0.f,0.f,0.f,0.f,0.f,0.f};
        v8f aU1 = {0.f,0.f,0.f,0.f,0.f,0.f,0.f,0.f};
        const unsigned short* grow = gb + ew + (size_t)(n0 + lrow) * D_DIM;
        const unsigned short* urow = ub + ew + (size_t)(n0 + lrow) * D_DIM;
        for (int d0 = 0; d0 < D_DIM; d0 += 32) {
            FragAB a0, a1, bg, bu;
            // A 16x32 bf16: vregs 0-3 hold K = half*8..+7, vregs 4-7 hold K = 16+half*8..+7
            a0.u[0] = *reinterpret_cast<const u32x4*>(xrow0 + d0 + half * 8);
            a0.u[1] = *reinterpret_cast<const u32x4*>(xrow0 + d0 + 16 + half * 8);
            a1.u[0] = *reinterpret_cast<const u32x4*>(xrow1 + d0 + half * 8);
            a1.u[1] = *reinterpret_cast<const u32x4*>(xrow1 + d0 + 16 + half * 8);
            // B 32x16 bf16: lane column n=lrow, vregs 0-3 hold K = half*8..+7, vregs 4-7 hold K = 16+half*8..+7
            bg.u[0] = *reinterpret_cast<const u32x4*>(grow + d0 + half * 8);
            bg.u[1] = *reinterpret_cast<const u32x4*>(grow + d0 + 16 + half * 8);
            bu.u[0] = *reinterpret_cast<const u32x4*>(urow + d0 + half * 8);
            bu.u[1] = *reinterpret_cast<const u32x4*>(urow + d0 + 16 + half * 8);
            // 4 WMMAs per 4 fragment loads: each B fragment reused twice
            aG0 = __builtin_amdgcn_wmma_f32_16x16x32_bf16(
                      false, a0.v, false, bg.v, (short)0, aG0, false, false);
            aG1 = __builtin_amdgcn_wmma_f32_16x16x32_bf16(
                      false, a1.v, false, bg.v, (short)0, aG1, false, false);
            aU0 = __builtin_amdgcn_wmma_f32_16x16x32_bf16(
                      false, a0.v, false, bu.v, (short)0, aU0, false, false);
            aU1 = __builtin_amdgcn_wmma_f32_16x16x32_bf16(
                      false, a1.v, false, bu.v, (short)0, aU1, false, false);
        }
        // h = silu(g) * u -> bf16 into LDS. D elem j: row m = j + half*8 (+16), col = n0 + lrow
        #pragma unroll
        for (int j = 0; j < 8; ++j) {
            int m = j + half * 8;
            s_h[m * I_DIM + n0 + lrow]        = f32_to_bf16(silu_mul(aG0[j], aU0[j]));
            s_h[(m + 16) * I_DIM + n0 + lrow] = f32_to_bf16(silu_mul(aG1[j], aU1[j]));
        }
    }
    __syncthreads();

    // ---------------- Phase B ----------------
    const unsigned short* hrow0 = s_h + lrow * I_DIM;
    const unsigned short* hrow1 = s_h + (lrow + 16) * I_DIM;
    for (int dt = 0; dt < 8; ++dt) {
        const int d0 = wv * 128 + dt * 16;
        v8f acc0 = {0.f,0.f,0.f,0.f,0.f,0.f,0.f,0.f};
        v8f acc1 = {0.f,0.f,0.f,0.f,0.f,0.f,0.f,0.f};
        const unsigned short* drow = db + ew + (size_t)(d0 + lrow) * I_DIM;
        for (int i0 = 0; i0 < I_DIM; i0 += 32) {
            FragAB a0, a1, b;
            a0.u[0] = *reinterpret_cast<const u32x4*>(hrow0 + i0 + half * 8);
            a0.u[1] = *reinterpret_cast<const u32x4*>(hrow0 + i0 + 16 + half * 8);
            a1.u[0] = *reinterpret_cast<const u32x4*>(hrow1 + i0 + half * 8);
            a1.u[1] = *reinterpret_cast<const u32x4*>(hrow1 + i0 + 16 + half * 8);
            b.u[0]  = *reinterpret_cast<const u32x4*>(drow + i0 + half * 8);
            b.u[1]  = *reinterpret_cast<const u32x4*>(drow + i0 + 16 + half * 8);
            acc0 = __builtin_amdgcn_wmma_f32_16x16x32_bf16(
                       false, a0.v, false, b.v, (short)0, acc0, false, false);
            acc1 = __builtin_amdgcn_wmma_f32_16x16x32_bf16(
                       false, a1.v, false, b.v, (short)0, acc1, false, false);
        }
        #pragma unroll
        for (int j = 0; j < 8; ++j) {
            int m = j + half * 8;
            int s0 = s_slot[m];
            int s1 = s_slot[m + 16];
            if (s0 >= 0)
                outslot[(size_t)s0 * D_DIM + d0 + lrow] = s_w[m] * acc0[j];
            if (s1 >= 0)
                outslot[(size_t)s1 * D_DIM + d0 + lrow] = s_w[m + 16] * acc1[j];
        }
    }
}

// ---------------------------------------------------------------- combine K slots
__global__ void combine_kernel(const float* __restrict__ os,
                               float* __restrict__ y, int n) {
    int i = blockIdx.x * blockDim.x + threadIdx.x;
    if (i >= n) return;
    int t = i >> 10;           // D = 1024
    int d = i & 1023;
    size_t base = (size_t)t * TOPK * D_DIM + d;
    y[i] = os[base] + os[base + D_DIM];
}

// ---------------------------------------------------------------- launch
extern "C" void kernel_launch(void* const* d_in, const int* in_sizes, int n_in,
                              void* d_out, int out_size, void* d_ws, size_t ws_size,
                              hipStream_t stream) {
    const float*         x          = (const float*)d_in[0];
    const unsigned char* token_mask = (const unsigned char*)d_in[1];
    const float*         weights    = (const float*)d_in[2];
    const int*           indices    = (const int*)d_in[3];
    const float*         gate       = (const float*)d_in[4];
    const float*         up         = (const float*)d_in[5];
    const float*         down       = (const float*)d_in[6];
    float*               y          = (float*)d_out;

    char* ws = (char*)d_ws;
    size_t off = 0;
    int*   counts = (int*)(ws + off);            off += 256;
    int*   lists  = (int*)(ws + off);            off += (size_t)N_EXP * RPE * 4;
    float* wl     = (float*)(ws + off);          off += (size_t)N_EXP * RPE * 4;
    unsigned short* xb = (unsigned short*)(ws + off); off += (size_t)T_TOK * D_DIM * 2;
    unsigned short* gb = (unsigned short*)(ws + off); off += (size_t)N_EXP * I_DIM * D_DIM * 2;
    unsigned short* ub = (unsigned short*)(ws + off); off += (size_t)N_EXP * I_DIM * D_DIM * 2;
    unsigned short* db = (unsigned short*)(ws + off); off += (size_t)N_EXP * D_DIM * I_DIM * 2;
    float* outslot = (float*)(ws + off);         off += (size_t)T_TOK * TOPK * D_DIM * 4;

    zero_counts_kernel<<<1, 32, 0, stream>>>(counts);

    const int nx4 = T_TOK * D_DIM / 4;
    const int nw4 = N_EXP * I_DIM * D_DIM / 4;
    cvt_kernel<<<1024, 256, 0, stream>>>(x,    xb, nx4);
    cvt_kernel<<<2048, 256, 0, stream>>>(gate, gb, nw4);
    cvt_kernel<<<2048, 256, 0, stream>>>(up,   ub, nw4);
    cvt_kernel<<<2048, 256, 0, stream>>>(down, db, nw4);

    route_kernel<<<(T_TOK + 255) / 256, 256, 0, stream>>>(
        indices, weights, token_mask, counts, lists, wl);

    dim3 grid(RPE / TILE_M, N_EXP);   // 128 tiles x 8 experts; empty tiles exit early
    moe_kernel<<<grid, 256, 0, stream>>>(xb, gb, ub, db, counts, lists, wl, outslot);

    combine_kernel<<<(T_TOK * D_DIM + 255) / 256, 256, 0, stream>>>(
        outslot, y, T_TOK * D_DIM);
}
